// ShifaMindPhase2GAT_42408507081344
// MI455X (gfx1250) — matrix operand order repaired and checked by reading, not compiled, not run on hardware
//
#include <hip/hip_runtime.h>
#include <hip/hip_bf16.h>
#include <cmath>

// ---------------------------------------------------------------------------
// CDNA5 / gfx1250 types for WMMA
// ---------------------------------------------------------------------------
typedef __bf16 bf16;
typedef __attribute__((ext_vector_type(16))) __bf16 v16bf;
typedef __attribute__((ext_vector_type(8)))  __bf16 v8bf;
typedef __attribute__((ext_vector_type(8)))  float  v8f;

__device__ __forceinline__ unsigned short f2bf_us(float f) {
    unsigned u = __float_as_uint(f);
    u = u + 0x7FFFu + ((u >> 16) & 1u);          // round-to-nearest-even
    return (unsigned short)(u >> 16);
}
__device__ __forceinline__ bf16 us2bf(unsigned short s) {
    bf16 r; __builtin_memcpy(&r, &s, 2); return r;
}
__device__ __forceinline__ bf16 f2bf(float f) { return us2bf(f2bf_us(f)); }
__device__ __forceinline__ float bf2f(bf16 h) {
    unsigned short s; __builtin_memcpy(&s, &h, 2);
    return __uint_as_float(((unsigned)s) << 16);
}

__device__ __forceinline__ float sigmoidf_(float x) { return 1.0f / (1.0f + expf(-x)); }

// float atomic max via int/uint ordering trick (native GLOBAL_ATOMIC_MAX/MIN)
__device__ __forceinline__ void atomicMaxF(float* a, float v) {
    if (v >= 0.0f) atomicMax((int*)a, __float_as_int(v));
    else           atomicMin((unsigned int*)a, __float_as_uint(v));
}

// ---------------------------------------------------------------------------
// WMMA GEMM, bf16 operands:  C[M,N] = act( alpha * A(MxK) * B(NxK)^T + bias )
//   A: bf16 row-major (lda elements).  B: bf16 (N,K) row-major (ldb elements).
//   K must be a multiple of 32; lda multiple of 8; ldb multiple of 8.
//   Out-of-range rows/cols are address-clamped (junk lands only in masked
//   store lanes) -> single branchless inner loop, EXEC stays all-1s.
//   Output: Ch!=null -> bf16 store, else fp32 to Cf.  act: 0 none, 1 tanh.
// One wave computes one 16x16 tile with v_wmma_f32_16x16x32_bf16.
// ---------------------------------------------------------------------------
__global__ void __launch_bounds__(128)
wmma_gemm(const bf16* __restrict__ A, const bf16* __restrict__ Bm,
          const float* __restrict__ bias, float* __restrict__ Cf,
          bf16* __restrict__ Ch,
          int M, int N, int K, int lda, int ldb, int ldc,
          float alpha, int act)
{
    const int lane = threadIdx.x;           // 0..31
    const int wave = threadIdx.y;           // 0..3
    const int m0   = (blockIdx.y * 4 + wave) * 16;
    const int n0   = blockIdx.x * 16;
    if (m0 >= M) return;                    // wave-uniform: EXEC stays all-1s

    const int li  = lane & 15;
    const int hi  = lane >> 4;
    const int am  = m0 + li;
    const int bn  = n0 + li;
    const int amc = (am < M) ? am : (M - 1);   // address clamp only
    const int bnc = (bn < N) ? bn : (N - 1);

    const bf16* Ar = A  + (size_t)amc * lda;
    const bf16* Br = Bm + (size_t)bnc * ldb;

    v8f acc = {};
    #pragma unroll 2
    for (int kt = 0; kt < K; kt += 32) {
        __builtin_prefetch(Ar + kt + 256, 0, 1);
        v8bf  a_lo = *(const v8bf*) (Ar + kt + 8 * hi);        // K = kt+8hi .. +7
        v8bf  a_hi = *(const v8bf*) (Ar + kt + 16 + 8 * hi);   // K = kt+16+8hi .. +7
        v8bf  b_lo = *(const v8bf*) (Br + kt + 8 * hi);        // K = kt+8hi .. +7
        v8bf  b_hi = *(const v8bf*) (Br + kt + 16 + 8 * hi);   // K = kt+16+8hi .. +7
        v16bf av, bv;
        #pragma unroll
        for (int j = 0; j < 8; ++j) { av[j] = a_lo[j]; av[8 + j] = a_hi[j]; bv[j] = b_lo[j]; bv[8 + j] = b_hi[j]; }
        acc = __builtin_amdgcn_wmma_f32_16x16x32_bf16(
                  false, av, false, bv, (short)0, acc, false, false);
    }

    // store: C/D layout (8 VGPRs, M split by lane-half)
    #pragma unroll
    for (int i = 0; i < 8; ++i) {
        int m = m0 + i + 8 * hi;
        if (m < M && bn < N) {
            float v = acc[i] * alpha + (bias ? bias[bn] : 0.0f);
            if (act == 1) v = tanhf(v);
            if (Ch) Ch[(size_t)m * ldc + bn] = f2bf(v);
            else    Cf[(size_t)m * ldc + bn] = v;
        }
    }
}

// ---------------------------------------------------------------------------
// Conversion / transpose kernels
// ---------------------------------------------------------------------------
// 8 floats -> 8 bf16 per thread (b128 in, b128 out)
__global__ void f2bf8_kernel(const float* __restrict__ s, bf16* __restrict__ d, long n8) {
    long i = blockIdx.x * (long)blockDim.x + threadIdx.x;
    if (i >= n8) return;
    const float4* sp = (const float4*)(s + i * 8);
    float4 x0 = sp[0], x1 = sp[1];
    union { unsigned short us[8]; uint4 u4; } pk;
    pk.us[0] = f2bf_us(x0.x); pk.us[1] = f2bf_us(x0.y);
    pk.us[2] = f2bf_us(x0.z); pk.us[3] = f2bf_us(x0.w);
    pk.us[4] = f2bf_us(x1.x); pk.us[5] = f2bf_us(x1.y);
    pk.us[6] = f2bf_us(x1.z); pk.us[7] = f2bf_us(x1.w);
    *(uint4*)(d + i * 8) = pk.u4;
}

// wt(N,K) = bf16( w(K,N)^T )
__global__ void wtrans_kernel(const float* __restrict__ w, bf16* __restrict__ wt, int Kd, int Nd) {
    int i = blockIdx.x * blockDim.x + threadIdx.x;
    if (i >= Kd * Nd) return;
    int n = i / Kd, k = i % Kd;
    wt[i] = f2bf(w[(size_t)k * Nd + n]);
}

// vvt[h][d][c(pad128)] = vvh[c][h*96+d], zero-padded c>=111
__global__ void vvt_kernel(const bf16* __restrict__ vvh, bf16* __restrict__ vvt) {
    int i = blockIdx.x * blockDim.x + threadIdx.x;
    if (i >= 8 * 96 * 128) return;
    int h = i / (96 * 128);
    int r = i % (96 * 128);
    int d = r / 128, c = r % 128;
    vvt[i] = (c < 111) ? vvh[(size_t)c * 768 + h * 96 + d] : us2bf(0);
}

// enrt[b][ch][s] = enrh[b][s][ch]
__global__ void enrt_kernel(const bf16* __restrict__ enrh, bf16* __restrict__ enrt) {
    long i = blockIdx.x * (long)blockDim.x + threadIdx.x;
    if (i >= 8L * 768 * 512) return;
    int b  = (int)(i / (768 * 512));
    int r  = (int)(i % (768 * 512));
    int ch = r / 512, s = r % 512;
    enrt[i] = enrh[((size_t)b * 512 + s) * 768 + ch];
}

// attnh[row][c(pad128)] = bf16(attn[row][c]), zero-padded c>=111
__global__ void pack_attn_kernel(const float* __restrict__ attn, bf16* __restrict__ attnh) {
    long i = blockIdx.x * (long)blockDim.x + threadIdx.x;
    if (i >= 32768L * 128) return;
    long row = i >> 7; int c = (int)(i & 127);
    attnh[i] = (c < 111) ? f2bf(attn[row * 111 + c]) : us2bf(0);
}

// ---------------------------------------------------------------------------
// Elementwise / reduction kernels
// ---------------------------------------------------------------------------
__global__ void fill_kernel(float* p, float v, long n) {
    long i = blockIdx.x * (long)blockDim.x + threadIdx.x;
    if (i < n) p[i] = v;
}

// x[i] = act(x[i] + bias[i % C]); optional bf16 mirror. act: 0 none, 1 elu
__global__ void bias_act_kernel(float* x, const float* bias, bf16* hout,
                                long total, int C, int act) {
    long i = blockIdx.x * (long)blockDim.x + threadIdx.x;
    if (i >= total) return;
    float v = x[i] + bias[i % C];
    if (act == 1) v = (v > 0.0f) ? v : (expf(v) - 1.0f);
    x[i] = v;
    if (hout) hout[i] = f2bf(v);
}

// asrc[n,h] = sum_c h[n,h,c]*a_src[h,c]; adst likewise
__global__ void gat_coef_kernel(const float* __restrict__ h,
                                const float* __restrict__ a_src,
                                const float* __restrict__ a_dst,
                                float* asrc, float* adst,
                                int N, int heads, int outch)
{
    long idx = blockIdx.x * (long)blockDim.x + threadIdx.x;
    if (idx >= (long)N * heads) return;
    int n = (int)(idx / heads), hd = (int)(idx % heads);
    const float* hp = h + (size_t)n * heads * outch + (size_t)hd * outch;
    const float* as = a_src + (size_t)hd * outch;
    const float* ad = a_dst + (size_t)hd * outch;
    float s = 0.0f, d = 0.0f;
    for (int c = 0; c < outch; ++c) { s += hp[c] * as[c]; d += hp[c] * ad[c]; }
    asrc[idx] = s; adst[idx] = d;
}

__device__ __forceinline__ void edge_sd(const int* ei, int e, int E, int& s, int& d) {
    if (e < E) { s = ei[e]; d = ei[E + e]; } else { s = d = e - E; }
}

__global__ void edge_max_kernel(const int* ei, const float* asrc, const float* adst,
                                float* mbuf, int E, int ET, int heads)
{
    long idx = blockIdx.x * (long)blockDim.x + threadIdx.x;
    if (idx >= (long)ET * heads) return;
    int e = (int)(idx / heads), hd = (int)(idx % heads);
    int s, d; edge_sd(ei, e, E, s, d);
    float x = asrc[(size_t)s * heads + hd] + adst[(size_t)d * heads + hd];
    x = (x > 0.0f) ? x : 0.2f * x;                         // leaky_relu 0.2
    atomicMaxF(&mbuf[(size_t)d * heads + hd], x);
}

__global__ void edge_sum_kernel(const int* ei, const float* asrc, const float* adst,
                                const float* mbuf, float* den, int E, int ET, int heads)
{
    long idx = blockIdx.x * (long)blockDim.x + threadIdx.x;
    if (idx >= (long)ET * heads) return;
    int e = (int)(idx / heads), hd = (int)(idx % heads);
    int s, d; edge_sd(ei, e, E, s, d);
    float x = asrc[(size_t)s * heads + hd] + adst[(size_t)d * heads + hd];
    x = (x > 0.0f) ? x : 0.2f * x;
    float p = expf(x - mbuf[(size_t)d * heads + hd]);
    atomicAdd(&den[(size_t)d * heads + hd], p);
}

__global__ void edge_scatter_kernel(const int* ei, const float* asrc, const float* adst,
                                    const float* mbuf, const float* den,
                                    const float* __restrict__ h, float* out,
                                    int E, int ET, int heads, int outch)
{
    long idx = blockIdx.x * (long)blockDim.x + threadIdx.x;
    if (idx >= (long)ET * heads) return;
    int e = (int)(idx / heads), hd = (int)(idx % heads);
    int s, d; edge_sd(ei, e, E, s, d);
    float x = asrc[(size_t)s * heads + hd] + adst[(size_t)d * heads + hd];
    x = (x > 0.0f) ? x : 0.2f * x;
    float p = expf(x - mbuf[(size_t)d * heads + hd]);
    float alpha = p / (den[(size_t)d * heads + hd] + 1e-16f);
    const float* hs = h + (size_t)s * heads * outch + (size_t)hd * outch;
    float* od = out + (size_t)d * heads * outch + (size_t)hd * outch;
    for (int c = 0; c < outch; ++c) atomicAdd(&od[c], hs[c] * alpha);
}

__global__ void gather_rows_kernel(const float* g, const int* idx, bf16* out) {
    int i = blockIdx.x * blockDim.x + threadIdx.x;
    if (i >= 111 * 256) return;
    int r = i / 256, c = i % 256;
    out[i] = f2bf(g[(size_t)idx[r] * 256 + c]);
}

__global__ void build_fused_kernel(const float* bert, const float* gatc, bf16* fused) {
    int i = blockIdx.x * blockDim.x + threadIdx.x;
    if (i >= 111 * 1536) return;
    int c = i / 1536, j = i % 1536;
    fused[i] = f2bf((j < 768) ? bert[(size_t)c * 768 + j]
                              : gatc[(size_t)c * 768 + (j - 768)]);
}

// row layernorm (+optional relu); one block per row; bf16 output
__global__ void ln_rows_kernel(const float* __restrict__ x, const float* g, const float* be,
                               bf16* y, int W, int relu)
{
    __shared__ float s1[256], s2[256];
    int row = blockIdx.x, tid = threadIdx.x;
    const float* xr = x + (size_t)row * W;
    float a = 0.0f, b = 0.0f;
    for (int j = tid; j < W; j += blockDim.x) { float v = xr[j]; a += v; b += v * v; }
    s1[tid] = a; s2[tid] = b; __syncthreads();
    for (int s = 128; s > 0; s >>= 1) {
        if (tid < s) { s1[tid] += s1[tid + s]; s2[tid] += s2[tid + s]; }
        __syncthreads();
    }
    float mean = s1[0] / W;
    float var  = s2[0] / W - mean * mean;
    float inv  = rsqrtf(var + 1e-5f);
    bf16* yr = y + (size_t)row * W;
    for (int j = tid; j < W; j += blockDim.x) {
        float v = (xr[j] - mean) * inv * g[j] + be[j];
        if (relu) v = fmaxf(v, 0.0f);
        yr[j] = f2bf(v);
    }
}

// enriched(bf16) = LN(hidden + sigmoid(gs)*context); also emits graph_strength
__global__ void add_ln_kernel(const float* __restrict__ hid, const float* __restrict__ ctx,
                              const float* gsc, const float* g, const float* be,
                              bf16* y, float* gs_out, int W)
{
    __shared__ float s1[256], s2[256];
    float gs = sigmoidf_(gsc[0]);
    if (blockIdx.x == 0 && threadIdx.x == 0) gs_out[0] = gs;
    int row = blockIdx.x, tid = threadIdx.x;
    const float* hr = hid + (size_t)row * W;
    const float* cr = ctx + (size_t)row * W;
    float a = 0.0f, b = 0.0f;
    for (int j = tid; j < W; j += blockDim.x) {
        float v = hr[j] + gs * cr[j]; a += v; b += v * v;
    }
    s1[tid] = a; s2[tid] = b; __syncthreads();
    for (int s = 128; s > 0; s >>= 1) {
        if (tid < s) { s1[tid] += s1[tid + s]; s2[tid] += s2[tid + s]; }
        __syncthreads();
    }
    float mean = s1[0] / W;
    float var  = s2[0] / W - mean * mean;
    float inv  = rsqrtf(var + 1e-5f);
    bf16* yr = y + (size_t)row * W;
    for (int j = tid; j < W; j += blockDim.x)
        yr[j] = f2bf((hr[j] + gs * cr[j] - mean) * inv * g[j] + be[j]);
}

// in-place softmax over last dim, one block per row
__global__ void softmax_rows_kernel(float* x, int W) {
    __shared__ float red[128];
    int row = blockIdx.x, tid = threadIdx.x;
    float* xr = x + (size_t)row * W;
    float m = -INFINITY;
    for (int j = tid; j < W; j += blockDim.x) m = fmaxf(m, xr[j]);
    red[tid] = m; __syncthreads();
    for (int s = 64; s > 0; s >>= 1) { if (tid < s) red[tid] = fmaxf(red[tid], red[tid + s]); __syncthreads(); }
    m = red[0]; __syncthreads();
    float sum = 0.0f;
    for (int j = tid; j < W; j += blockDim.x) sum += expf(xr[j] - m);
    red[tid] = sum; __syncthreads();
    for (int s = 64; s > 0; s >>= 1) { if (tid < s) red[tid] += red[tid + s]; __syncthreads(); }
    sum = red[0];
    for (int j = tid; j < W; j += blockDim.x) xr[j] = expf(xr[j] - m) / sum;
}

// attn_weights[b,s,c] = mean_h attn[b,h,s,c]
__global__ void attn_mean_kernel(const float* __restrict__ attn, float* out) {
    long i = blockIdx.x * (long)blockDim.x + threadIdx.x;
    const long total = 8L * 512 * 111;
    if (i >= total) return;
    int b = (int)(i / (512 * 111));
    int r = (int)(i % (512 * 111));
    int s = r / 111, c = r % 111;
    float acc = 0.0f;
    for (int h = 0; h < 8; ++h)
        acc += attn[(((size_t)(b * 8 + h)) * 512 + s) * 111 + c];
    out[i] = acc * 0.125f;
}

// masked softmax over S (axis=1) per (b,l); in-place on att (B,S,L)
__global__ void laat_softmax_kernel(float* att, const int* mask, int S, int L) {
    __shared__ float red[128];
    int b = blockIdx.x / L, l = blockIdx.x % L, tid = threadIdx.x;
    float m = -INFINITY;
    for (int s = tid; s < S; s += blockDim.x) {
        float v = (mask[(size_t)b * S + s] == 0) ? -INFINITY : att[((size_t)b * S + s) * L + l];
        m = fmaxf(m, v);
    }
    red[tid] = m; __syncthreads();
    for (int s = 64; s > 0; s >>= 1) { if (tid < s) red[tid] = fmaxf(red[tid], red[tid + s]); __syncthreads(); }
    m = red[0]; __syncthreads();
    float sum = 0.0f;
    for (int s = tid; s < S; s += blockDim.x) {
        float v = (mask[(size_t)b * S + s] == 0) ? -INFINITY : att[((size_t)b * S + s) * L + l];
        sum += expf(v - m);
    }
    red[tid] = sum; __syncthreads();
    for (int s = 64; s > 0; s >>= 1) { if (tid < s) red[tid] += red[tid + s]; __syncthreads(); }
    sum = red[0];
    for (int s = tid; s < S; s += blockDim.x) {
        float v = (mask[(size_t)b * S + s] == 0) ? -INFINITY : att[((size_t)b * S + s) * L + l];
        att[((size_t)b * S + s) * L + l] = expf(v - m) / sum;
    }
}

// wT[b,l,s] = bf16(att[b,s,l])
__global__ void transpose_w_kernel(const float* att, bf16* wT) {
    long i = blockIdx.x * (long)blockDim.x + threadIdx.x;
    if (i >= 8L * 50 * 512) return;
    int b = (int)(i / (50 * 512));
    int r = (int)(i % (50 * 512));
    int l = r / 512, s = r % 512;
    wT[i] = f2bf(att[((size_t)b * 512 + s) * 50 + l]);
}

__global__ void concept_scores_kernel(const float* cl, float* cs) {
    int i = blockIdx.x * blockDim.x + threadIdx.x;
    if (i >= 8 * 111) return;
    cs[i] = sigmoidf_(cl[i]);
}

__global__ void final_logits_kernel(const float* __restrict__ lreps, const float* cscores,
                                    const float* ow, const float* ob, const float* c2l,
                                    float* logits)
{
    int i = blockIdx.x * blockDim.x + threadIdx.x;
    if (i >= 8 * 50) return;
    int b = i / 50, l = i % 50;
    float acc = ob[0];
    const float* lr = lreps + ((size_t)b * 50 + l) * 768;
    for (int h = 0; h < 768; ++h) acc += lr[h] * ow[h];
    for (int c = 0; c < 111; ++c) acc += cscores[(size_t)b * 111 + c] * c2l[(size_t)l * 111 + c];
    logits[i] = acc;
}

// ---------------------------------------------------------------------------
// Host orchestration
// ---------------------------------------------------------------------------
extern "C" void kernel_launch(void* const* d_in, const int* in_sizes, int n_in,
                              void* d_out, int out_size, void* d_ws, size_t ws_size,
                              hipStream_t stream)
{
    (void)in_sizes; (void)n_in; (void)out_size; (void)ws_size;

    const float* hidden   = (const float*)d_in[0];
    const int*   amask    = (const int*)  d_in[1];
    const float* bert     = (const float*)d_in[2];
    const float* graph_x  = (const float*)d_in[3];
    const int*   ei       = (const int*)  d_in[4];
    const int*   cnidx    = (const int*)  d_in[5];
    const float* w1       = (const float*)d_in[6];
    const float* asrc1    = (const float*)d_in[7];
    const float* adst1    = (const float*)d_in[8];
    const float* bias1    = (const float*)d_in[9];
    const float* w2       = (const float*)d_in[10];
    const float* asrc2    = (const float*)d_in[11];
    const float* adst2    = (const float*)d_in[12];
    const float* bias2    = (const float*)d_in[13];
    const float* gproj_w  = (const float*)d_in[14];
    const float* gproj_b  = (const float*)d_in[15];
    const float* fus_w    = (const float*)d_in[16];
    const float* fus_b    = (const float*)d_in[17];
    const float* fus_ln_g = (const float*)d_in[18];
    const float* fus_ln_b = (const float*)d_in[19];
    const float* inp_w    = (const float*)d_in[20];
    const float* inp_b    = (const float*)d_in[21];
    const float* outp_w   = (const float*)d_in[22];
    const float* outp_b   = (const float*)d_in[23];
    const float* ln_g     = (const float*)d_in[24];
    const float* ln_b     = (const float*)d_in[25];
    const float* chead_w  = (const float*)d_in[26];
    const float* chead_b  = (const float*)d_in[27];
    const float* lfirst_w = (const float*)d_in[28];
    const float* lsecond_w= (const float*)d_in[29];
    const float* lout_w   = (const float*)d_in[30];
    const float* lout_b   = (const float*)d_in[31];
    const float* c2l_w    = (const float*)d_in[32];
    const float* gscale   = (const float*)d_in[33];

    const int N = 50000, E = 800000, ET = E + N;

    // ---- output layout (concatenated, return order) ----
    float* out       = (float*)d_out;
    float* o_logits  = out;                          // 8*50
    float* o_clogit  = out + 400;                    // 8*111
    float* o_cscore  = out + 1288;                   // 8*111
    float* o_lreps   = out + 2176;                   // 8*50*768
    float* o_attnw   = out + 2176 + 307200;          // 8*512*111
    float* o_gs      = out + 2176 + 307200 + 454656; // 1

    // ---- workspace bump allocator (byte based, 16B aligned) ----
    char* wsb = (char*)d_ws;
    size_t off = 0;
    auto allocF = [&](size_t n) { float* p = (float*)(wsb + off); off += ((n * 4 + 15) & ~(size_t)15); return p; };
    auto allocH = [&](size_t n) { bf16*  p = (bf16*) (wsb + off); off += ((n * 2 + 15) & ~(size_t)15); return p; };

    // fp32 scratch
    float* H    = allocF((size_t)N * 256);     // h1/h2, later context(fp32)
    float* G    = allocF((size_t)N * 256);     // gat outs, later enh(fp32)
    float* ASRC = allocF((size_t)N * 4);
    float* ADST = allocF((size_t)N * 4);
    float* MB   = allocF((size_t)N * 4);
    float* DEN  = allocF((size_t)N * 4);
    float* GATC = allocF(111 * 768);
    float* ATTN = allocF((size_t)8 * 8 * 512 * 111);
    float* ATT  = allocF((size_t)4096 * 50);
    // bf16 operands
    bf16* XH    = allocH((size_t)N * 256);
    bf16* GH    = allocH((size_t)N * 256);
    bf16* W1H   = allocH(256 * 256);
    bf16* W2H   = allocH(256 * 256);
    bf16* GSELH = allocH(111 * 256);
    bf16* GPROJH= allocH(768 * 256);
    bf16* FUSEDH= allocH(111 * 1536);
    bf16* FUSWH = allocH(768 * 1536);
    bf16* ENHH  = allocH(111 * 768);
    bf16* INPWH = allocH((size_t)2304 * 768);
    bf16* HIDH  = allocH((size_t)4096 * 768);
    bf16* QH    = allocH((size_t)4096 * 768);
    bf16* KKH   = allocH(111 * 768);
    bf16* VVH   = allocH(111 * 768);
    bf16* VVTH  = allocH(8 * 96 * 128);
    bf16* ATTNH = allocH((size_t)32768 * 128);
    bf16* CTXH  = allocH((size_t)4096 * 768);
    bf16* OUTPWH= allocH(768 * 768);
    bf16* ENRH  = allocH((size_t)4096 * 768);
    bf16* LF_H  = allocH(768 * 768);
    bf16* UH    = allocH((size_t)4096 * 768);
    bf16* LS_H  = allocH(50 * 768);
    bf16* WTH   = allocH((size_t)8 * 50 * 512);
    bf16* ENRT  = allocH((size_t)8 * 768 * 512);
    bf16* CHEADH= allocH(111 * 768);

    float* ENHF = G;    // reuse: fp32 fusion pre-LN (G free after gather)
    float* CTXF = H;    // reuse: fp32 context (H free after GAT)

    auto gemm = [&](const bf16* A, const bf16* Bm, const float* bias,
                    float* Cf, bf16* Ch,
                    int M, int Nn, int K, int lda, int ldb, int ldc,
                    float alpha, int act) {
        dim3 blk(32, 4);
        dim3 grd((Nn + 15) / 16, (M + 63) / 64);
        wmma_gemm<<<grd, blk, 0, stream>>>(A, Bm, bias, Cf, Ch, M, Nn, K,
                                           lda, ldb, ldc, alpha, act);
    };
    auto fill = [&](float* p, float v, long n) {
        fill_kernel<<<(unsigned)((n + 255) / 256), 256, 0, stream>>>(p, v, n);
    };
    auto conv8 = [&](const float* s, bf16* d, long n) {  // n elements (mult of 8)
        f2bf8_kernel<<<(unsigned)((n / 8 + 255) / 256), 256, 0, stream>>>(s, d, n / 8);
    };

    // ---- operand conversions (weights & inputs, once) ----
    conv8(graph_x, XH, (long)N * 256);
    wtrans_kernel<<<(256 * 256 + 255) / 256, 256, 0, stream>>>(w1, W1H, 256, 256);
    wtrans_kernel<<<(256 * 256 + 255) / 256, 256, 0, stream>>>(w2, W2H, 256, 256);
    conv8(gproj_w, GPROJH, 768 * 256);
    conv8(fus_w, FUSWH, 768 * 1536);
    conv8(inp_w, INPWH, (long)2304 * 768);
    conv8(hidden, HIDH, (long)4096 * 768);
    conv8(outp_w, OUTPWH, 768 * 768);
    conv8(lfirst_w, LF_H, 768 * 768);
    conv8(lsecond_w, LS_H, 50 * 768);
    conv8(chead_w, CHEADH, 111 * 768);

    // ================= GAT layer 1 (heads=4, outch=64, concat) =================
    gemm(XH, W1H, nullptr, H, nullptr, N, 256, 256, 256, 256, 256, 1.0f, 0);
    gat_coef_kernel<<<(N * 4 + 255) / 256, 256, 0, stream>>>(H, asrc1, adst1, ASRC, ADST, N, 4, 64);
    fill(MB, -INFINITY, (long)N * 4);
    fill(DEN, 0.0f, (long)N * 4);
    fill(G, 0.0f, (long)N * 256);
    {
        long t = (long)ET * 4; unsigned g = (unsigned)((t + 255) / 256);
        edge_max_kernel    <<<g, 256, 0, stream>>>(ei, ASRC, ADST, MB, E, ET, 4);
        edge_sum_kernel    <<<g, 256, 0, stream>>>(ei, ASRC, ADST, MB, DEN, E, ET, 4);
        edge_scatter_kernel<<<g, 256, 0, stream>>>(ei, ASRC, ADST, MB, DEN, H, G, E, ET, 4, 64);
    }
    bias_act_kernel<<<(unsigned)(((long)N * 256 + 255) / 256), 256, 0, stream>>>(
        G, bias1, GH, (long)N * 256, 256, 1);   // elu; bf16 mirror for next GEMM

    // ================= GAT layer 2 (heads=1, outch=256) =================
    gemm(GH, W2H, nullptr, H, nullptr, N, 256, 256, 256, 256, 256, 1.0f, 0);
    gat_coef_kernel<<<(N + 255) / 256, 256, 0, stream>>>(H, asrc2, adst2, ASRC, ADST, N, 1, 256);
    fill(MB, -INFINITY, N);
    fill(DEN, 0.0f, N);
    fill(G, 0.0f, (long)N * 256);
    {
        long t = (long)ET; unsigned g = (unsigned)((t + 255) / 256);
        edge_max_kernel    <<<g, 256, 0, stream>>>(ei, ASRC, ADST, MB, E, ET, 1);
        edge_sum_kernel    <<<g, 256, 0, stream>>>(ei, ASRC, ADST, MB, DEN, E, ET, 1);
        edge_scatter_kernel<<<g, 256, 0, stream>>>(ei, ASRC, ADST, MB, DEN, H, G, E, ET, 1, 256);
    }
    bias_act_kernel<<<(unsigned)(((long)N * 256 + 255) / 256), 256, 0, stream>>>(
        G, bias2, nullptr, (long)N * 256, 256, 0);

    // ========== concept projection + fusion (batch-invariant: compute once) ==========
    gather_rows_kernel<<<(111 * 256 + 255) / 256, 256, 0, stream>>>(G, cnidx, GSELH);
    gemm(GSELH, GPROJH, gproj_b, GATC, nullptr, 111, 768, 256, 256, 256, 768, 1.0f, 0);
    build_fused_kernel<<<(111 * 1536 + 255) / 256, 256, 0, stream>>>(bert, GATC, FUSEDH);
    gemm(FUSEDH, FUSWH, fus_b, ENHF, nullptr, 111, 768, 1536, 1536, 1536, 768, 1.0f, 0);
    ln_rows_kernel<<<111, 256, 0, stream>>>(ENHF, fus_ln_g, fus_ln_b, ENHH, 768, 1);

    // k, v (once), q (per token) -- bf16 outputs feed the next GEMMs
    gemm(ENHH, INPWH + (size_t)768 * 768,  inp_b + 768,  nullptr, KKH, 111, 768, 768, 768, 768, 768, 1.0f, 0);
    gemm(ENHH, INPWH + (size_t)1536 * 768, inp_b + 1536, nullptr, VVH, 111, 768, 768, 768, 768, 768, 1.0f, 0);
    gemm(HIDH, INPWH, inp_b, nullptr, QH, 4096, 768, 768, 768, 768, 768, 1.0f, 0);

    // ========== cross-attention ==========
    const float sc = 1.0f / sqrtf(96.0f);
    for (int b = 0; b < 8; ++b)
        for (int h = 0; h < 8; ++h)
            gemm(QH + (size_t)b * 512 * 768 + h * 96, KKH + h * 96, nullptr,
                 ATTN + (size_t)(b * 8 + h) * 512 * 111, nullptr,
                 512, 111, 96, 768, 768, 111, sc, 0);
    softmax_rows_kernel<<<8 * 8 * 512, 128, 0, stream>>>(ATTN, 111);
    attn_mean_kernel<<<(unsigned)((8L * 512 * 111 + 255) / 256), 256, 0, stream>>>(ATTN, o_attnw);
    pack_attn_kernel<<<(unsigned)((32768L * 128 + 255) / 256), 256, 0, stream>>>(ATTN, ATTNH);
    vvt_kernel<<<(8 * 96 * 128 + 255) / 256, 256, 0, stream>>>(VVH, VVTH);
    for (int b = 0; b < 8; ++b)
        for (int h = 0; h < 8; ++h)
            gemm(ATTNH + (size_t)(b * 8 + h) * 512 * 128, VVTH + (size_t)h * 96 * 128, nullptr,
                 nullptr, CTXH + (size_t)b * 512 * 768 + h * 96,
                 512, 96, 128, 128, 128, 768, 1.0f, 0);
    gemm(CTXH, OUTPWH, outp_b, CTXF, nullptr, 4096, 768, 768, 768, 768, 768, 1.0f, 0);

    // enriched(bf16) = LN(hidden + sigmoid(gscale)*context); emit graph_strength
    add_ln_kernel<<<4096, 256, 0, stream>>>(hidden, CTXF, gscale, ln_g, ln_b, ENRH, o_gs, 768);

    // ========== LAAT ==========
    gemm(ENRH, LF_H, nullptr, nullptr, UH, 4096, 768, 768, 768, 768, 768, 1.0f, 1); // U = tanh
    gemm(UH, LS_H, nullptr, ATT, nullptr, 4096, 50, 768, 768, 768, 50, 1.0f, 0);
    laat_softmax_kernel<<<8 * 50, 128, 0, stream>>>(ATT, amask, 512, 50);
    transpose_w_kernel<<<(unsigned)((8L * 50 * 512 + 255) / 256), 256, 0, stream>>>(ATT, WTH);
    enrt_kernel<<<(unsigned)((8L * 768 * 512 + 255) / 256), 256, 0, stream>>>(ENRH, ENRT);
    for (int b = 0; b < 8; ++b)
        gemm(WTH + (size_t)b * 50 * 512, ENRT + (size_t)b * 768 * 512, nullptr,
             o_lreps + (size_t)b * 50 * 768, nullptr,
             50, 768, 512, 512, 512, 768, 1.0f, 0);

    // ========== concept head + final logits ==========
    gemm(HIDH, CHEADH, chead_b, o_clogit, nullptr, 8, 111, 768, 512 * 768, 768, 111, 1.0f, 0);
    concept_scores_kernel<<<(8 * 111 + 255) / 256, 256, 0, stream>>>(o_clogit, o_cscore);
    final_logits_kernel<<<(8 * 50 + 255) / 256, 256, 0, stream>>>(
        o_lreps, o_cscore, lout_w, lout_b, c2l_w, o_logits);
}
